// PointTransformerSemanticSegmentation_51651276702290
// MI455X (gfx1250) — matrix operand order repaired and checked by reading, not compiled, not run on hardware
//
#include <hip/hip_runtime.h>
#include <hip/hip_bf16.h>
#include <stddef.h>
#include <stdint.h>

// ---------------------------------------------------------------------------
// Types for WMMA fragments (CDNA5 / gfx1250, wave32)
// ---------------------------------------------------------------------------
typedef __bf16 bf16_t;
typedef bf16_t v16bf __attribute__((ext_vector_type(16)));
typedef float  v8f   __attribute__((ext_vector_type(8)));

// ---------------------------------------------------------------------------
// Weight repack: W[K,N] fp32 -> Wp bf16 in fragment-linear order
//   Wp[((tn*KC + kc)*32 + lane)*16 + i]  =  W[kc*32 + (i<8 ? half*8 + i : 16 + half*8 + i-8), tn*16 + l16]
// (zero-padded for ragged K and N). This makes every B-fragment load in the
// GEMM a single contiguous 32-byte vector load per lane.
// ---------------------------------------------------------------------------
__global__ void repack_w_kernel(const float* __restrict__ W, bf16_t* __restrict__ Wp,
                                int K, int N, int KC, int tilesNpad)
{
    const size_t t = (size_t)blockIdx.x * blockDim.x + threadIdx.x;
    const size_t total = (size_t)tilesNpad * KC * 32 * 16;
    if (t >= total) return;
    const int i    = (int)(t & 15);
    const int lane = (int)((t >> 4) & 31);
    size_t r = t >> 9;
    const int kc = (int)(r % KC);
    const int tn = (int)(r / KC);
    const int half = lane >> 4, l16 = lane & 15;
    const int kb  = kc * 32 + (i < 8 ? half * 8 + i : 16 + half * 8 + (i - 8));
    const int col = tn * 16 + l16;
    const float v = (kb < K && col < N) ? W[(size_t)kb * N + col] : 0.0f;
    Wp[t] = (bf16_t)v;
}

// ---------------------------------------------------------------------------
// Fast GEMM: out[M,N] = A[M,Kpad] @ W + bias, optional ReLU.
// Requirements: M % 16 == 0, Kpad % 32 == 0 (A physically padded with zeros),
// Wp pre-repacked (handles ragged true-K/N). Branch-free inner loop:
//   A: 4x global_load_b128 per lane (rows always valid), packed cvt to bf16
//   B: 2x contiguous 32B vector loads per lane from Wp
//   2x v_wmma_f32_16x16x32_bf16 per K-step (16x32 output tile per wave)
// Fragment layouts per CDNA5 ISA 7.12.2:
//   A 16x32: lane half 0 -> K {0..7,16..23}, half 1 -> K {8..15,24..31}, M=lane%16
//   B 32x16: lane half 0 -> K {0..7,16..23}, half 1 -> K {8..15,24..31}, N=lane%16
//   C/D 16x16: VGPR r -> M = r + 8*half, N = lane%16
// ---------------------------------------------------------------------------
__global__ __launch_bounds__(256)
void gemm_fast_kernel(const float* __restrict__ A, const bf16_t* __restrict__ Wp,
                      const float* __restrict__ bias, float* __restrict__ out,
                      int M, int Kpad, int N, int relu)
{
    const int lane    = threadIdx.x & 31;
    const int wave    = threadIdx.x >> 5;
    const int tilesNP = (N + 31) >> 5;            // pairs of 16-wide N tiles
    const int tiles   = (M >> 4) * tilesNP;
    const int tile    = blockIdx.x * 8 + wave;
    if (tile >= tiles) return;                     // wave-uniform; EXEC stays full

    const int tm  = tile / tilesNP;
    const int tnp = tile % tilesNP;
    const int tn0 = tnp * 2, tn1 = tn0 + 1;

    const int half  = lane >> 4;
    const int l16   = lane & 15;
    const int kbase = half * 8;                    // A-fragment K interleave base
    const int arow  = tm * 16 + l16;               // always < M

    const int KC = Kpad >> 5;
    const float*  ap  = A  + (size_t)arow * Kpad + kbase;
    const bf16_t* wp0 = Wp + (((size_t)tn0 * KC) * 32 + lane) * 16;
    const bf16_t* wp1 = Wp + (((size_t)tn1 * KC) * 32 + lane) * 16;

    v8f acc0 = {}, acc1 = {};
    for (int kc = 0; kc < KC; kc++) {
        const float* a_ = ap + (size_t)kc * 32;
        __builtin_prefetch(a_ + 32, 0, 1);         // speculative: next K chunk
        const float4 f0 = *(const float4*)(a_ + 0);
        const float4 f1 = *(const float4*)(a_ + 4);
        const float4 f2 = *(const float4*)(a_ + 16);
        const float4 f3 = *(const float4*)(a_ + 20);
        v16bf a;
        a[0]  = (bf16_t)f0.x; a[1]  = (bf16_t)f0.y; a[2]  = (bf16_t)f0.z; a[3]  = (bf16_t)f0.w;
        a[4]  = (bf16_t)f1.x; a[5]  = (bf16_t)f1.y; a[6]  = (bf16_t)f1.z; a[7]  = (bf16_t)f1.w;
        a[8]  = (bf16_t)f2.x; a[9]  = (bf16_t)f2.y; a[10] = (bf16_t)f2.z; a[11] = (bf16_t)f2.w;
        a[12] = (bf16_t)f3.x; a[13] = (bf16_t)f3.y; a[14] = (bf16_t)f3.z; a[15] = (bf16_t)f3.w;

        const v16bf b0 = *(const v16bf*)(wp0 + (size_t)kc * 512);
        const v16bf b1 = *(const v16bf*)(wp1 + (size_t)kc * 512);

        acc0 = __builtin_amdgcn_wmma_f32_16x16x32_bf16(false, a, false, b0,
                                                       (short)0, acc0, false, false);
        acc1 = __builtin_amdgcn_wmma_f32_16x16x32_bf16(false, a, false, b1,
                                                       (short)0, acc1, false, false);
    }

    const int col0 = tn0 * 16 + l16;
    const int col1 = tn1 * 16 + l16;
    const float b0v = (col0 < N) ? bias[col0] : 0.0f;
    const float b1v = (col1 < N) ? bias[col1] : 0.0f;
#pragma unroll
    for (int r = 0; r < 8; r++) {
        const int row = tm * 16 + r + half * 8;    // always < M
        if (col0 < N) {
            float v = acc0[r] + b0v;
            if (relu) v = fmaxf(v, 0.0f);
            out[(size_t)row * N + col0] = v;
        }
        if (col1 < N) {
            float v = acc1[r] + b1v;
            if (relu) v = fmaxf(v, 0.0f);
            out[(size_t)row * N + col1] = v;
        }
    }
}

// ---------------------------------------------------------------------------
// Brute-force 16-NN: one thread per query, fully unrolled register insertion
// sort (branchless cndmask chains, wave32-friendly).
// ---------------------------------------------------------------------------
__global__ void knn16_kernel(const float* __restrict__ qp, const float* __restrict__ cp,
                             int* __restrict__ idx, int B, int Nq, int Nc)
{
    const int t = blockIdx.x * blockDim.x + threadIdx.x;
    if (t >= B * Nq) return;
    const int b = t / Nq, n = t % Nq;
    const float* q = qp + ((size_t)b * Nq + n) * 3;
    const float qx = q[0], qy = q[1], qz = q[2];
    const float* cb = cp + (size_t)b * Nc * 3;

    float bd[16]; int bi[16];
#pragma unroll
    for (int j = 0; j < 16; j++) { bd[j] = 3.4e38f; bi[j] = 0; }

    for (int c = 0; c < Nc; c++) {
        const float dx = qx - cb[c * 3 + 0];
        const float dy = qy - cb[c * 3 + 1];
        const float dz = qz - cb[c * 3 + 2];
        float d = dx * dx + dy * dy + dz * dz;
        int  ii = c;
#pragma unroll
        for (int j = 0; j < 16; j++) {
            const bool sw = d < bd[j];
            const float td = bd[j]; const int ti = bi[j];
            bd[j] = sw ? d : td;  bi[j] = sw ? ii : ti;
            d     = sw ? td : d;  ii    = sw ? ti : ii;
        }
    }
    int* o = idx + (size_t)t * 16;
#pragma unroll
    for (int j = 0; j < 16; j++) o[j] = bi[j];
}

// gather: dst[b,m,k,c] = src[b, idx[b,m,k], c]
__global__ void gather_kernel(const float* __restrict__ src, const int* __restrict__ idx,
                              float* __restrict__ dst, int B, int Nsrc, int M, int d)
{
    const size_t t = (size_t)blockIdx.x * blockDim.x + threadIdx.x;
    const size_t total = (size_t)B * M * 16 * d;
    if (t >= total) return;
    const int c = (int)(t % d);
    size_t r = t / d;
    const int k = (int)(r % 16); r /= 16;
    const int m = (int)(r % M);
    const int b = (int)(r / M);
    const int id = idx[((size_t)b * M + m) * 16 + k];
    dst[t] = src[((size_t)b * Nsrc + id) * d + c];
}

// padded pos-diff: dst[b,n,k,c(0..31)] = c<3 ? pos[b,n,c]-pos[b,idx,c] : 0
__global__ void posdiff_pad_kernel(const float* __restrict__ pos, const int* __restrict__ idx,
                                   float* __restrict__ dst, int B, int N)
{
    const size_t t = (size_t)blockIdx.x * blockDim.x + threadIdx.x;
    const size_t total = (size_t)B * N * 16 * 32;
    if (t >= total) return;
    const int c = (int)(t % 32);
    size_t r = t / 32;
    const int k = (int)(r % 16); r /= 16;
    const int n = (int)(r % N);
    const int b = (int)(r / N);
    float v = 0.0f;
    if (c < 3) {
        const int id = idx[((size_t)b * N + n) * 16 + k];
        v = pos[((size_t)b * N + n) * 3 + c] - pos[((size_t)b * N + id) * 3 + c];
    }
    dst[t] = v;
}

// pad input features [M,din] -> [M,32] zero padded
__global__ void pad_x_kernel(const float* __restrict__ src, float* __restrict__ dst,
                             size_t M, int din)
{
    const size_t t = (size_t)blockIdx.x * blockDim.x + threadIdx.x;
    if (t >= M * 32) return;
    const int c = (int)(t % 32);
    const size_t m = t / 32;
    dst[t] = (c < din) ? src[m * din + c] : 0.0f;
}

// per-channel softmax over 16 neighbors: out = sum_k softmax_k(Q-K+pe) * (V+pe)
__global__ void attn_kernel(const float* __restrict__ Q, const float* __restrict__ Kf,
                            const float* __restrict__ Vf, const float* __restrict__ pe,
                            float* __restrict__ out, int BN, int d)
{
    const size_t t = (size_t)blockIdx.x * blockDim.x + threadIdx.x;
    if (t >= (size_t)BN * d) return;
    const size_t n = t / d;
    const int   c = (int)(t % d);
    const float q = Q[t];
    const size_t base = (n * 16) * (size_t)d + c;

    float s[16], w[16];
    float m = -3.4e38f;
#pragma unroll
    for (int k = 0; k < 16; k++) {
        const float pv = pe[base + (size_t)k * d];
        s[k] = q - Kf[base + (size_t)k * d] + pv;
        w[k] = Vf[base + (size_t)k * d] + pv;
        m = fmaxf(m, s[k]);
    }
    float sum = 0.0f;
#pragma unroll
    for (int k = 0; k < 16; k++) { s[k] = __expf(s[k] - m); sum += s[k]; }
    float acc = 0.0f;
#pragma unroll
    for (int k = 0; k < 16; k++) acc += s[k] * w[k];
    out[t] = acc / sum;
}

// x = relu(x * (gamma/sqrt(1+eps)) + beta)
__global__ void bnrelu_kernel(float* __restrict__ x, const float* __restrict__ g,
                              const float* __restrict__ bt, size_t M, int N)
{
    const size_t t = (size_t)blockIdx.x * blockDim.x + threadIdx.x;
    if (t >= M * (size_t)N) return;
    const int c = (int)(t % N);
    const float scale = g[c] * 0.99999500003749969f;   // 1/sqrt(1 + 1e-5)
    x[t] = fmaxf(x[t] * scale + bt[c], 0.0f);
}

__global__ void add_kernel(float* __restrict__ a, const float* __restrict__ b, size_t n)
{
    const size_t t = (size_t)blockIdx.x * blockDim.x + threadIdx.x;
    if (t < n) a[t] += b[t];
}

// max over 16 neighbors: src[M,16,d] -> dst[M,d]
__global__ void maxpool_kernel(const float* __restrict__ src, float* __restrict__ dst,
                               size_t Mrows, int d)
{
    const size_t t = (size_t)blockIdx.x * blockDim.x + threadIdx.x;
    if (t >= Mrows * (size_t)d) return;
    const size_t m = t / d;
    const int   c = (int)(t % d);
    const float* p = src + (m * 16) * (size_t)d + c;
    float v = p[0];
#pragma unroll
    for (int k = 1; k < 16; k++) v = fmaxf(v, p[(size_t)k * d]);
    dst[t] = v;
}

// p2 = pos[:, :n2]
__global__ void slicepos_kernel(const float* __restrict__ src, float* __restrict__ dst,
                                int B, int N, int n2)
{
    const size_t t = (size_t)blockIdx.x * blockDim.x + threadIdx.x;
    const size_t total = (size_t)B * n2 * 3;
    if (t >= total) return;
    const int c = (int)(t % 3);
    size_t r = t / 3;
    const int n = (int)(r % n2);
    const int b = (int)(r / n2);
    dst[t] = src[((size_t)b * N + n) * 3 + c];
}

// 3-NN + normalized inverse-distance weights
__global__ void knn3w_kernel(const float* __restrict__ dq, const float* __restrict__ sp,
                             int* __restrict__ idx3, float* __restrict__ w3,
                             int B, int Nd, int Ns)
{
    const int t = blockIdx.x * blockDim.x + threadIdx.x;
    if (t >= B * Nd) return;
    const int b = t / Nd, n = t % Nd;
    const float* q = dq + ((size_t)b * Nd + n) * 3;
    const float qx = q[0], qy = q[1], qz = q[2];
    const float* cb = sp + (size_t)b * Ns * 3;

    float bd[3] = {3.4e38f, 3.4e38f, 3.4e38f};
    int   bi[3] = {0, 0, 0};
    for (int c = 0; c < Ns; c++) {
        const float dx = qx - cb[c * 3 + 0];
        const float dy = qy - cb[c * 3 + 1];
        const float dz = qz - cb[c * 3 + 2];
        float d = dx * dx + dy * dy + dz * dz;
        int  ii = c;
#pragma unroll
        for (int j = 0; j < 3; j++) {
            const bool sw = d < bd[j];
            const float td = bd[j]; const int ti = bi[j];
            bd[j] = sw ? d : td;  bi[j] = sw ? ii : ti;
            d     = sw ? td : d;  ii    = sw ? ti : ii;
        }
    }
    float w[3], sum = 0.0f;
#pragma unroll
    for (int j = 0; j < 3; j++) { w[j] = 1.0f / (sqrtf(bd[j]) + 1e-8f); sum += w[j]; }
    const size_t base = (size_t)t * 3;
#pragma unroll
    for (int j = 0; j < 3; j++) { idx3[base + j] = bi[j]; w3[base + j] = w[j] / sum; }
}

// out[b,n,c] = sum_j w3[b,n,j] * xs[b, idx3[b,n,j], c]
__global__ void interp_kernel(const float* __restrict__ xs, const int* __restrict__ idx3,
                              const float* __restrict__ w3, float* __restrict__ out,
                              int B, int Nd, int Ns, int d)
{
    const size_t t = (size_t)blockIdx.x * blockDim.x + threadIdx.x;
    const size_t total = (size_t)B * Nd * d;
    if (t >= total) return;
    const int c = (int)(t % d);
    size_t r = t / d;
    const int n = (int)(r % Nd);
    const int b = (int)(r / Nd);
    const size_t base = ((size_t)b * Nd + n) * 3;
    float acc = 0.0f;
#pragma unroll
    for (int j = 0; j < 3; j++)
        acc += w3[base + j] * xs[((size_t)b * Ns + idx3[base + j]) * d + c];
    out[t] = acc;
}

// ---------------------------------------------------------------------------
// Host-side orchestration
// ---------------------------------------------------------------------------
struct Lin   { const float *w, *b; };
struct Proj  { Lin l1, l2; };
struct Ptb   { Proj lin1, q, k, v, pos, lin2; };
struct Trans { Lin l1; const float *gamma, *beta; Lin l2; };

struct Scratch {
    int*    idx;                        // [B*4096*16]
    float  *G1, *G2, *G3, *G4, *G5;     // [B*4096*16*32] each (reused per level)
    float*  PDP;                        // padded pos-diff [B*4096*16*32]
    float  *T1, *T2, *T3;               // [B*4096*32] each
    int*    idx3;  float* w3;           // [B*4096*3]
    bf16_t* Wp;                         // repacked weights (max 512x512 padded)
};

static inline unsigned ewg(size_t n) { return (unsigned)((n + 255) / 256); }

// Kpad: physical row stride of A (multiple of 32, zero padded). Ktrue: logical K of W.
static void run_gemm(hipStream_t st, const float* A, const float* W, const float* bias,
                     float* out, int M, int Kpad, int Ktrue, int N, int relu, bf16_t* Wp)
{
    const int KC = Kpad >> 5;
    const int tilesNP = (N + 31) >> 5;
    const size_t rp_total = (size_t)tilesNP * 2 * KC * 512;
    repack_w_kernel<<<ewg(rp_total), 256, 0, st>>>(W, Wp, Ktrue, N, KC, tilesNP * 2);
    const long tiles = (long)(M >> 4) * tilesNP;
    gemm_fast_kernel<<<(unsigned)((tiles + 7) / 8), 256, 0, st>>>(A, Wp, bias, out, M, Kpad, N, relu);
}

static void run_proj(hipStream_t st, const Proj& pr, const float* A, float* tmp,
                     float* out, int M, int din, int dout, bf16_t* Wp)
{
    run_gemm(st, A,   pr.l1.w, pr.l1.b, tmp, M, din,  din,  dout, 1, Wp);
    run_gemm(st, tmp, pr.l2.w, pr.l2.b, out, M, dout, dout, dout, 0, Wp);
}

// pt_block: x (in/out, [B,N,d]) updated in place with residual
static void pt_block_run(hipStream_t st, const Ptb& pr, float* x, const float* pos,
                         int B, int N, int d, const Scratch& S)
{
    const int M = B * N, Mk = M * 16;
    knn16_kernel<<<ewg((size_t)M), 256, 0, st>>>(pos, pos, S.idx, B, N, N);
    run_proj(st, pr.lin1, x, S.T3, S.T1, M, d, d, S.Wp);                 // x1 = T1
    gather_kernel<<<ewg((size_t)Mk * d), 256, 0, st>>>(S.T1, S.idx, S.G1, B, N, N, d);
    posdiff_pad_kernel<<<ewg((size_t)Mk * 32), 256, 0, st>>>(pos, S.idx, S.PDP, B, N);
    run_proj(st, pr.q, S.T1, S.T3, S.T2, M,  d, d, S.Wp);                // Q  = T2
    run_proj(st, pr.k, S.G1, S.G5, S.G2, Mk, d, d, S.Wp);                // K  = G2
    run_proj(st, pr.v, S.G1, S.G5, S.G3, Mk, d, d, S.Wp);                // V  = G3
    // positional projection: A padded to K=32, true K=3
    run_gemm(st, S.PDP, pr.pos.l1.w, pr.pos.l1.b, S.G5, Mk, 32, 3, d, 1, S.Wp);
    run_gemm(st, S.G5,  pr.pos.l2.w, pr.pos.l2.b, S.G4, Mk, d,  d, d, 0, S.Wp);
    attn_kernel<<<ewg((size_t)M * d), 256, 0, st>>>(S.T2, S.G2, S.G3, S.G4, S.T3, M, d);
    run_proj(st, pr.lin2, S.T3, S.T2, S.T1, M, d, d, S.Wp);
    add_kernel<<<ewg((size_t)M * d), 256, 0, st>>>(x, S.T1, (size_t)M * d);
}

static void tr_down_run(hipStream_t st, const Trans& tr, const float* x, const float* pos,
                        int B, int N, int n2, int din, int dout,
                        float* xd, float* pd, const Scratch& S)
{
    slicepos_kernel<<<ewg((size_t)B * n2 * 3), 256, 0, st>>>(pos, pd, B, N, n2);
    knn16_kernel<<<ewg((size_t)B * n2), 256, 0, st>>>(pd, pos, S.idx, B, n2, N);
    gather_kernel<<<ewg((size_t)B * n2 * 16 * din), 256, 0, st>>>(x, S.idx, S.G1, B, N, n2, din);
    const int Mk = B * n2 * 16;
    run_gemm(st, S.G1, tr.l1.w, tr.l1.b, S.G5, Mk, din, din, dout, 0, S.Wp);
    bnrelu_kernel<<<ewg((size_t)Mk * dout), 256, 0, st>>>(S.G5, tr.gamma, tr.beta, (size_t)Mk, dout);
    run_gemm(st, S.G5, tr.l2.w, tr.l2.b, S.G2, Mk, dout, dout, dout, 0, S.Wp);
    maxpool_kernel<<<ewg((size_t)B * n2 * dout), 256, 0, st>>>(S.G2, xd, (size_t)B * n2, dout);
}

static void tr_up_run(hipStream_t st, const Trans& tr, const float* xs,
                      const float* ps, const float* pdense,
                      int B, int ns, int nd, int din, int dout,
                      float* out, const Scratch& S)
{
    const int Ms = B * ns;
    run_gemm(st, xs, tr.l1.w, tr.l1.b, S.T3, Ms, din, din, dout, 0, S.Wp);
    bnrelu_kernel<<<ewg((size_t)Ms * dout), 256, 0, st>>>(S.T3, tr.gamma, tr.beta, (size_t)Ms, dout);
    run_gemm(st, S.T3, tr.l2.w, tr.l2.b, S.T2, Ms, dout, dout, dout, 0, S.Wp);
    knn3w_kernel<<<ewg((size_t)B * nd), 256, 0, st>>>(pdense, ps, S.idx3, S.w3, B, nd, ns);
    interp_kernel<<<ewg((size_t)B * nd * dout), 256, 0, st>>>(S.T2, S.idx3, S.w3, out, B, nd, ns, dout);
}

// --- parameter unpacking in make_params() insertion order ---
static Lin  getLin (void* const* d, int& i) { Lin  l; l.w = (const float*)d[i++]; l.b = (const float*)d[i++]; return l; }
static Proj getProj(void* const* d, int& i) { Proj p; p.l1 = getLin(d, i); p.l2 = getLin(d, i); return p; }
static Ptb  getPtb (void* const* d, int& i) { Ptb t; t.lin1 = getProj(d, i); t.q = getProj(d, i); t.k = getProj(d, i);
                                              t.v = getProj(d, i); t.pos = getProj(d, i); t.lin2 = getProj(d, i); return t; }
static Trans getTrans(void* const* d, int& i) { Trans t; t.l1 = getLin(d, i);
                                                t.gamma = (const float*)d[i++]; t.beta = (const float*)d[i++];
                                                t.l2 = getLin(d, i); return t; }

struct Arena {
    char* base; size_t off;
    void* take(size_t bytes) { size_t a = (off + 255) & ~(size_t)255; off = a + bytes; return base + a; }
    float* f(size_t n) { return (float*)take(n * sizeof(float)); }
    int*   i(size_t n) { return (int*)take(n * sizeof(int)); }
};

extern "C" void kernel_launch(void* const* d_in, const int* in_sizes, int n_in,
                              void* d_out, int out_size, void* d_ws, size_t ws_size,
                              hipStream_t stream)
{
    const int B = 2, N0 = 4096;
    const float* x_in = (const float*)d_in[0];   // [B, 4096, 6]
    const float* p    = (const float*)d_in[1];   // [B, 4096, 3]

    int pi = 2;
    const Lin   linear_in  = getLin (d_in, pi);
    const Ptb   pt_in      = getPtb (d_in, pi);
    const Trans tr_down1   = getTrans(d_in, pi);  const Ptb pt_down1 = getPtb(d_in, pi);
    const Trans tr_down2   = getTrans(d_in, pi);  const Ptb pt_down2 = getPtb(d_in, pi);
    const Trans tr_down3   = getTrans(d_in, pi);  const Ptb pt_down3 = getPtb(d_in, pi);
    const Trans tr_down4   = getTrans(d_in, pi);  const Ptb pt_down4 = getPtb(d_in, pi);
    const Lin   linear_mid = getLin (d_in, pi);
    const Trans tr_up4     = getTrans(d_in, pi);  const Ptb pt_up4   = getPtb(d_in, pi);
    const Trans tr_up3     = getTrans(d_in, pi);  const Ptb pt_up3   = getPtb(d_in, pi);
    const Trans tr_up2     = getTrans(d_in, pi);  const Ptb pt_up2   = getPtb(d_in, pi);
    const Trans tr_up1     = getTrans(d_in, pi);  const Ptb pt_up1   = getPtb(d_in, pi);
    const Lin   linear_cls = getLin (d_in, pi);

    Arena ar{(char*)d_ws, 0};
    Scratch S;
    const size_t GBIG = (size_t)B * 4096 * 16 * 32;   // 4.19M floats
    S.idx = ar.i((size_t)B * 4096 * 16);
    S.G1 = ar.f(GBIG); S.G2 = ar.f(GBIG); S.G3 = ar.f(GBIG);
    S.G4 = ar.f(GBIG); S.G5 = ar.f(GBIG);
    S.PDP = ar.f(GBIG);
    S.T1 = ar.f((size_t)B * 4096 * 32);
    S.T2 = ar.f((size_t)B * 4096 * 32);
    S.T3 = ar.f((size_t)B * 4096 * 32);
    S.idx3 = ar.i((size_t)B * 4096 * 3);
    S.w3   = ar.f((size_t)B * 4096 * 3);
    S.Wp   = (bf16_t*)ar.take(512 * 512 * 2 * sizeof(bf16_t));  // padded max

    float* XP   = ar.f((size_t)B * 4096 * 32);  // x_in padded 6->32
    float* xin  = ar.f((size_t)B * 4096 * 32);
    float* xd1  = ar.f((size_t)B * 1024 * 64);
    float* xd2  = ar.f((size_t)B * 256 * 128);
    float* xd3  = ar.f((size_t)B * 64 * 256);
    float* xd4  = ar.f((size_t)B * 16 * 512);
    float* pd1  = ar.f((size_t)B * 1024 * 3);
    float* pd2  = ar.f((size_t)B * 256 * 3);
    float* pd3  = ar.f((size_t)B * 64 * 3);
    float* pd4  = ar.f((size_t)B * 16 * 3);
    float* xmid = ar.f((size_t)B * 16 * 512);
    float* xu4  = ar.f((size_t)B * 64 * 256);
    float* xu3  = ar.f((size_t)B * 256 * 128);
    float* xu2  = ar.f((size_t)B * 1024 * 64);
    float* xu1  = ar.f((size_t)B * 4096 * 32);

    // ---------------- encoder ----------------
    pad_x_kernel<<<ewg((size_t)B * N0 * 32), 256, 0, stream>>>(x_in, XP, (size_t)B * N0, 6);
    run_gemm(stream, XP, linear_in.w, linear_in.b, xin, B * N0, 32, 6, 32, 0, S.Wp);
    pt_block_run(stream, pt_in, xin, p, B, N0, 32, S);

    tr_down_run(stream, tr_down1, xin, p,   B, 4096, 1024, 32,  64,  xd1, pd1, S);
    pt_block_run(stream, pt_down1, xd1, pd1, B, 1024, 64, S);
    tr_down_run(stream, tr_down2, xd1, pd1, B, 1024, 256,  64,  128, xd2, pd2, S);
    pt_block_run(stream, pt_down2, xd2, pd2, B, 256, 128, S);
    tr_down_run(stream, tr_down3, xd2, pd2, B, 256,  64,   128, 256, xd3, pd3, S);
    pt_block_run(stream, pt_down3, xd3, pd3, B, 64, 256, S);
    tr_down_run(stream, tr_down4, xd3, pd3, B, 64,   16,   256, 512, xd4, pd4, S);
    pt_block_run(stream, pt_down4, xd4, pd4, B, 16, 512, S);

    // ---------------- bottleneck ----------------
    run_gemm(stream, xd4, linear_mid.w, linear_mid.b, xmid, B * 16, 512, 512, 512, 0, S.Wp);

    // ---------------- decoder ----------------
    tr_up_run(stream, tr_up4, xmid, pd4, pd3, B, 16,   64,   512, 256, xu4, S);
    add_kernel<<<ewg((size_t)B * 64 * 256), 256, 0, stream>>>(xu4, xd3, (size_t)B * 64 * 256);
    pt_block_run(stream, pt_up4, xu4, pd3, B, 64, 256, S);

    tr_up_run(stream, tr_up3, xu4, pd3, pd2, B, 64,   256,  256, 128, xu3, S);
    add_kernel<<<ewg((size_t)B * 256 * 128), 256, 0, stream>>>(xu3, xd2, (size_t)B * 256 * 128);
    pt_block_run(stream, pt_up3, xu3, pd2, B, 256, 128, S);

    tr_up_run(stream, tr_up2, xu3, pd2, pd1, B, 256,  1024, 128, 64,  xu2, S);
    add_kernel<<<ewg((size_t)B * 1024 * 64), 256, 0, stream>>>(xu2, xd1, (size_t)B * 1024 * 64);
    pt_block_run(stream, pt_up2, xu2, pd1, B, 1024, 64, S);

    tr_up_run(stream, tr_up1, xu2, pd1, p,   B, 1024, 4096, 64,  32,  xu1, S);
    add_kernel<<<ewg((size_t)B * 4096 * 32), 256, 0, stream>>>(xu1, xin, (size_t)B * 4096 * 32);
    pt_block_run(stream, pt_up1, xu1, p, B, 4096, 32, S);

    // ---------------- classifier ----------------
    run_gemm(stream, xu1, linear_cls.w, linear_cls.b, (float*)d_out, B * 4096, 32, 32, 13, 0, S.Wp);
}
